// GCN_more_layer_SGD_58248346468477
// MI455X (gfx1250) — matrix operand order repaired and checked by reading, not compiled, not run on hardware
//
#include <hip/hip_runtime.h>
#include <hip/hip_bf16.h>
#include <math.h>

// ---------------------------------------------------------------------------
// GCN 3-layer forward for MI455X (gfx1250, wave32).
//   GEMMs:  bf16 WMMA (v_wmma_f32_16x16x32_bf16), f32 accumulate,
//           LDS-staged tiles in fragment-friendly (transposed-W) layout.
//   Edge aggregation: bandwidth-bound gather + global_atomic_add_f32.
// ---------------------------------------------------------------------------

typedef __bf16 bf16_t;
typedef __attribute__((ext_vector_type(16))) __bf16 v16bf;
typedef __attribute__((ext_vector_type(8)))  __bf16 v8bf;
typedef __attribute__((ext_vector_type(8)))  float  v8f;

// ---------------------------------------------------------------------------
// Utility kernels
// ---------------------------------------------------------------------------
__global__ void zero_f32(float* __restrict__ p, long long n) {
    long long i = (long long)blockIdx.x * blockDim.x + threadIdx.x;
    if (i < n) p[i] = 0.0f;
}

__global__ void deg_init(float* __restrict__ deg, int n) {
    int i = blockIdx.x * blockDim.x + threadIdx.x;
    if (i < n) deg[i] = 1.0f;              // self-loop contribution
}

__global__ void deg_count(const long long* __restrict__ dst,
                          float* __restrict__ deg, long long E) {
    long long e = (long long)blockIdx.x * blockDim.x + threadIdx.x;
    if (e < E) atomicAdd(&deg[dst[e]], 1.0f);
}

__global__ void deg_rsqrt_inplace(float* __restrict__ d, int n) {
    int i = blockIdx.x * blockDim.x + threadIdx.x;
    if (i < n) d[i] = rsqrtf(d[i]);        // deg -> deg^{-1/2}
}

// ---------------------------------------------------------------------------
// WMMA GEMM:  Out[N x H] = A[N x K] * W[K x H]   (row-major, fp32 in/out,
// bf16 multiply, f32 accumulate).
// Block = (H/16) waves; block computes a 64-row x H-col stripe
// (4 M-subtiles per wave; B fragment reused across the 4 WMMAs).
// LDS layout: W stored transposed with padded row stride S = K+8 bf16 so
// every lane's 16-element B run is contiguous + 16B aligned -> ds_load_b128.
// ---------------------------------------------------------------------------
template <int K, int H>
__global__ __launch_bounds__(2 * H)
void gemm_bf16_wmma(const float* __restrict__ A,
                    const float* __restrict__ W,
                    float* __restrict__ Out, int N) {
    constexpr int S  = K + 8;   // padded LDS row stride (bf16 elems, mult of 8)
    constexpr int MT = 4;       // 16-row subtiles per block  => 64 rows/block
    __shared__ __align__(16) bf16_t lwt[H * S];        // W^T : [n][k]
    __shared__ __align__(16) bf16_t la[(16 * MT) * S]; // A   : [row][k]

    const int tid  = threadIdx.x;
    const int row0 = blockIdx.x * (16 * MT);

    // Stage W transposed (coalesced global read, scattered LDS write)
    for (int idx = tid; idx < K * H; idx += blockDim.x) {
        int k = idx / H, n = idx - k * H;
        lwt[n * S + k] = (bf16_t)W[idx];
    }
    // Stage A rows with clamp (keeps all lanes valid for WMMA)
    for (int idx = tid; idx < 16 * MT * K; idx += blockDim.x) {
        int r = idx / K, c = idx - r * K;
        int rr = row0 + r;
        if (rr >= N) rr = N - 1;
        la[r * S + c] = (bf16_t)A[(long long)rr * K + c];
    }
    __syncthreads();

    const int wave  = tid >> 5;        // column tile index
    const int lane  = tid & 31;
    const int n0    = wave * 16;
    const int lm    = lane & 15;       // A row / B col within tile
    const int khalf = lane >> 4;

    v8f acc[MT] = {};
    for (int kb = 0; kb < K; kb += 32) {
        // B 32x16 fragment: lane n=n0+lm, K run {khalf*16 .. +15} contiguous
        const v8bf* bp = (const v8bf*)&lwt[(n0 + lm) * S + kb + khalf * 16];
        v16bf b = __builtin_shufflevector(bp[0], bp[1],
                    0, 1, 2, 3, 4, 5, 6, 7, 8, 9, 10, 11, 12, 13, 14, 15);
        #pragma unroll
        for (int mt = 0; mt < MT; ++mt) {
            // A 16x32 fragment: row m, K run {khalf*16 .. +15} contiguous
            const bf16_t* ar = &la[(mt * 16 + lm) * S + kb + khalf * 16];
            v8bf alo = *(const v8bf*)ar;
            v8bf ahi = *(const v8bf*)(ar + 8);
            v16bf a = __builtin_shufflevector(alo, ahi,
                        0, 1, 2, 3, 4, 5, 6, 7, 8, 9, 10, 11, 12, 13, 14, 15);
            acc[mt] = __builtin_amdgcn_wmma_f32_16x16x32_bf16(
                false, a, false, b, (short)0, acc[mt], false, false);
        }
    }

    // C/D layout: lane -> N = n0+lm ; vgpr r -> M = rowbase + khalf*8 + r
    #pragma unroll
    for (int mt = 0; mt < MT; ++mt) {
        #pragma unroll
        for (int r = 0; r < 8; ++r) {
            int m = row0 + mt * 16 + khalf * 8 + r;
            if (m < N) Out[(long long)m * H + n0 + lm] = acc[mt][r];
        }
    }
}

// ---------------------------------------------------------------------------
// Edge aggregation: agg[dst] += h[src] * dinv[src]*dinv[dst]
// LPE lanes per edge, 4 floats (float4) per lane => H = 4*LPE.
// ---------------------------------------------------------------------------
template <int H, int LPE>
__global__ void edge_aggregate(const float* __restrict__ h,
                               const long long* __restrict__ src,
                               const long long* __restrict__ dst,
                               const float* __restrict__ dinv,
                               float* __restrict__ agg, long long E) {
    long long t = (long long)blockIdx.x * blockDim.x + threadIdx.x;
    long long e = t / LPE;
    int l = (int)(t - e * LPE);
    if (e >= E) return;

    if (l == 0) {   // stream the index arrays ahead (global_prefetch_b8)
        long long pe = e + 4096;
        if (pe < E) {
            __builtin_prefetch(&src[pe], 0, 0);
            __builtin_prefetch(&dst[pe], 0, 0);
        }
    }

    long long s = src[e], d = dst[e];
    float norm = dinv[s] * dinv[d];
    float4 v = ((const float4*)(h + s * H))[l];
    float* out = agg + d * H + (long long)l * 4;
    atomicAdd(out + 0, v.x * norm);
    atomicAdd(out + 1, v.y * norm);
    atomicAdd(out + 2, v.z * norm);
    atomicAdd(out + 3, v.w * norm);
}

// H == 1 specialization (one thread per edge)
__global__ void edge_aggregate1(const float* __restrict__ h,
                                const long long* __restrict__ src,
                                const long long* __restrict__ dst,
                                const float* __restrict__ dinv,
                                float* __restrict__ agg, long long E) {
    long long e = (long long)blockIdx.x * blockDim.x + threadIdx.x;
    if (e >= E) return;
    long long s = src[e], d = dst[e];
    atomicAdd(&agg[d], h[s] * dinv[s] * dinv[d]);
}

// ---------------------------------------------------------------------------
// Post: out = act(agg + hlin * dinv^2 + bias)   (elementwise)
// ACT: 0 = relu, 1 = sigmoid
// ---------------------------------------------------------------------------
template <int H, int ACT>
__global__ void post_act(const float* __restrict__ agg,
                         const float* __restrict__ hlin,
                         const float* __restrict__ dinv,
                         const float* __restrict__ bias,
                         float* __restrict__ out, int N) {
    long long idx = (long long)blockIdx.x * blockDim.x + threadIdx.x;
    long long total = (long long)N * H;
    if (idx >= total) return;
    long long node = idx / H;
    int f = (int)(idx - node * H);
    float di = dinv[node];
    float v = agg[idx] + hlin[idx] * di * di + bias[f];
    out[idx] = (ACT == 0) ? fmaxf(v, 0.0f) : 1.0f / (1.0f + __expf(-v));
}

// ---------------------------------------------------------------------------
// Layer-3 linear: out[i] = dot(h[i, 0:64], W3[0:64])   (F_OUT = 1)
// ---------------------------------------------------------------------------
__global__ void gemv64(const float* __restrict__ h,
                       const float* __restrict__ w,
                       float* __restrict__ out, int N) {
    int i = blockIdx.x * blockDim.x + threadIdx.x;
    if (i >= N) return;
    const float4* hp = (const float4*)(h + (long long)i * 64);
    const float4* wp = (const float4*)w;
    float acc = 0.0f;
    #pragma unroll
    for (int j = 0; j < 16; ++j) {
        float4 a = hp[j], b = wp[j];
        acc += a.x * b.x + a.y * b.y + a.z * b.z + a.w * b.w;
    }
    out[i] = acc;
}

// ---------------------------------------------------------------------------
// Launcher
// ---------------------------------------------------------------------------
extern "C" void kernel_launch(void* const* d_in, const int* in_sizes, int n_in,
                              void* d_out, int out_size, void* d_ws, size_t ws_size,
                              hipStream_t stream) {
    const float*     x   = (const float*)d_in[0];          // [N,128]
    const long long* ei  = (const long long*)d_in[1];      // [2,E] int64
    const float*     W1  = (const float*)d_in[2];          // [128,128]
    const float*     b1  = (const float*)d_in[3];
    const float*     W2  = (const float*)d_in[4];          // [128,64]
    const float*     b2  = (const float*)d_in[5];
    const float*     W3  = (const float*)d_in[6];          // [64,1]
    const float*     b3  = (const float*)d_in[7];

    const int       N   = in_sizes[0] / 128;
    const long long E   = (long long)in_sizes[1] / 2;
    const long long* src = ei;
    const long long* dst = ei + E;

    // Workspace: bufA | bufB (N*128 f32 each) | dinv (N f32)  ~= 51.4 MB
    float* bufA = (float*)d_ws;
    float* bufB = bufA + (size_t)N * 128;
    float* dinv = bufB + (size_t)N * 128;

    const int T = 256;
    auto blk  = [&](long long n) { return (unsigned)((n + T - 1) / T); };

    // deg^{-1/2}  (recomputed every call: replay-safe)
    deg_init<<<blk(N), T, 0, stream>>>(dinv, N);
    deg_count<<<blk(E), T, 0, stream>>>(dst, dinv, E);
    deg_rsqrt_inplace<<<blk(N), T, 0, stream>>>(dinv, N);

    const unsigned gemmBlocks = (unsigned)((N + 63) / 64);  // 64 rows / block

    // ---- Layer 1: K=128 -> H=128 ----
    gemm_bf16_wmma<128, 128><<<gemmBlocks, 256, 0, stream>>>(x, W1, bufA, N);
    zero_f32<<<blk((long long)N * 128), T, 0, stream>>>(bufB, (long long)N * 128);
    edge_aggregate<128, 32><<<blk(E * 32), T, 0, stream>>>(bufA, src, dst, dinv, bufB, E);
    post_act<128, 0><<<blk((long long)N * 128), T, 0, stream>>>(bufB, bufA, dinv, b1, bufB, N);

    // ---- Layer 2: K=128 -> H=64 ----
    gemm_bf16_wmma<128, 64><<<gemmBlocks, 128, 0, stream>>>(bufB, W2, bufA, N);
    zero_f32<<<blk((long long)N * 64), T, 0, stream>>>(bufB, (long long)N * 64);
    edge_aggregate<64, 16><<<blk(E * 16), T, 0, stream>>>(bufA, src, dst, dinv, bufB, E);
    post_act<64, 0><<<blk((long long)N * 64), T, 0, stream>>>(bufB, bufA, dinv, b2, bufB, N);

    // ---- Layer 3: K=64 -> 1 ----
    gemv64<<<blk(N), T, 0, stream>>>(bufB, W3, bufA, N);
    zero_f32<<<blk(N), T, 0, stream>>>(bufB, N);
    edge_aggregate1<<<blk(E), T, 0, stream>>>(bufA, src, dst, dinv, bufB, E);
    post_act<1, 1><<<blk(N), T, 0, stream>>>(bufB, bufA, dinv, b3, (float*)d_out, N);
}
